// CrossFormerBlock_13649406067175
// MI455X (gfx1250) — matrix operand order repaired and checked by reading, not compiled, not run on hardware
//
#include <hip/hip_runtime.h>
#include <hip/hip_bf16.h>
#include <math.h>

// ---------------------------------------------------------------------------
// CrossFormer block for gfx1250 (MI455X).  All GEMMs + attention einsums use
// v_wmma_f32_16x16x32_f16 (f16 in, fp32 accumulate).  LayerNorm / softmax /
// pos-bias MLP stay fp32.  All fragment loads are explicit 128-bit (int4)
// loads.  Workspace requirement: ~342 MB.
// ---------------------------------------------------------------------------

typedef __attribute__((ext_vector_type(16))) _Float16 v16h;
typedef __attribute__((ext_vector_type(8)))  _Float16 v8h;
typedef __attribute__((ext_vector_type(8)))  float    v8f;

#define N_TOK   65536        // B*H*W
#define DIM_C   512
#define N_HEAD  16
#define H_DIM   32
#define N_GRP   1024         // B * 64
#define GSIZE   64           // tokens per group
#define SCALE_Q 0.17677669529663687f   // 32^-0.5

// ---- workspace layout (byte offsets) --------------------------------------
#define H16_OFF   0ull                          // 64 MB  ln1/ln2 f16 activations
#define Q16_OFF   67108864ull                   // 64 MB
#define K16_OFF   134217728ull                  // 64 MB
#define V16_OFF   201326592ull                  // 64 MB
#define O16_OFF   268435456ull                  // 64 MB attention output (grouped)
#define MID16_OFF Q16_OFF                       // 256 MB MLP mid, reuses Q..O
#define P225_OFF  335544320ull                  // 14.4 KB pos-bias table
#define QKVW_OFF  (P225_OFF + 16384ull)         // 1.5 MB
#define PROJW_OFF (QKVW_OFF + 1572864ull)       // 0.5 MB
#define W1_OFF    (PROJW_OFF + 524288ull)       // 2 MB
#define W2_OFF    (W1_OFF + 2097152ull)         // 2 MB

// ---- 128-bit load helpers --------------------------------------------------
__device__ __forceinline__ v8h ld8h(const _Float16* p) {
  return __builtin_bit_cast(v8h, *(const int4*)p);   // global_load_b128 / ds_load_b128
}
__device__ __forceinline__ v16h mk16(v8h lo, v8h hi) {
  return __builtin_shufflevector(lo, hi, 0, 1, 2, 3, 4, 5, 6, 7,
                                 8, 9, 10, 11, 12, 13, 14, 15);
}
__device__ __forceinline__ v8f wmma16(v16h a, v16h b, v8f c) {
  return __builtin_amdgcn_wmma_f32_16x16x32_f16(false, a, false, b,
                                                (short)0, c, false, false);
}

// ---------------------------------------------------------------------------
// f32 -> f16 elementwise (weight conversion)
// ---------------------------------------------------------------------------
__global__ void cvt_f16_kernel(const float* __restrict__ in,
                               _Float16* __restrict__ out, int n) {
  int i = blockIdx.x * blockDim.x + threadIdx.x;
  if (i < n) out[i] = (_Float16)in[i];
}

// ---------------------------------------------------------------------------
// LayerNorm over 512 columns, one wave (32 lanes) per row, f16 output
// ---------------------------------------------------------------------------
__global__ __launch_bounds__(256)
void ln_f16_kernel(const float* __restrict__ x, const float* __restrict__ g,
                   const float* __restrict__ b, _Float16* __restrict__ out,
                   int rows) {
  int wv   = blockIdx.x * (blockDim.x >> 5) + (threadIdx.x >> 5);
  int lane = threadIdx.x & 31;
  if (wv >= rows) return;
  const float4* row = (const float4*)(x + (size_t)wv * DIM_C);
  float4 d[4];
  float s = 0.f, s2 = 0.f;
#pragma unroll
  for (int i = 0; i < 4; ++i) {
    d[i] = row[lane + 32 * i];
    s  += d[i].x + d[i].y + d[i].z + d[i].w;
    s2 += d[i].x * d[i].x + d[i].y * d[i].y + d[i].z * d[i].z + d[i].w * d[i].w;
  }
#pragma unroll
  for (int off = 16; off > 0; off >>= 1) {
    s  += __shfl_xor(s,  off, 32);
    s2 += __shfl_xor(s2, off, 32);
  }
  float mean = s * (1.f / DIM_C);
  float var  = s2 * (1.f / DIM_C) - mean * mean;
  float rstd = rsqrtf(var + 1e-5f);
  _Float16* o = out + (size_t)wv * DIM_C;
#pragma unroll
  for (int i = 0; i < 4; ++i) {
    int c0 = (lane + 32 * i) * 4;
    float v0 = (d[i].x - mean) * rstd * g[c0 + 0] + b[c0 + 0];
    float v1 = (d[i].y - mean) * rstd * g[c0 + 1] + b[c0 + 1];
    float v2 = (d[i].z - mean) * rstd * g[c0 + 2] + b[c0 + 2];
    float v3 = (d[i].w - mean) * rstd * g[c0 + 3] + b[c0 + 3];
    o[c0 + 0] = (_Float16)v0; o[c0 + 1] = (_Float16)v1;
    o[c0 + 2] = (_Float16)v2; o[c0 + 3] = (_Float16)v3;
  }
}

// ---------------------------------------------------------------------------
// Dynamic position-bias MLP on the 15x15=225 relative-coordinate table.
// ---------------------------------------------------------------------------
__device__ inline void ln_relu32(float* v, const float* g, const float* b) {
  float m = 0.f;
#pragma unroll
  for (int j = 0; j < 32; ++j) m += v[j];
  m *= (1.f / 32.f);
  float var = 0.f;
#pragma unroll
  for (int j = 0; j < 32; ++j) { float d = v[j] - m; var += d * d; }
  var *= (1.f / 32.f);
  float rs = rsqrtf(var + 1e-5f);
#pragma unroll
  for (int j = 0; j < 32; ++j) {
    float t = (v[j] - m) * rs * g[j] + b[j];
    v[j] = t > 0.f ? t : 0.f;
  }
}

__global__ void pos_mlp_kernel(
    const float* __restrict__ w0, const float* __restrict__ b0,
    const float* __restrict__ g0, const float* __restrict__ bb0,
    const float* __restrict__ w1, const float* __restrict__ b1,
    const float* __restrict__ g1, const float* __restrict__ bb1,
    const float* __restrict__ w2, const float* __restrict__ b2,
    const float* __restrict__ g2, const float* __restrict__ bb2,
    const float* __restrict__ w3, const float* __restrict__ b3,
    float* __restrict__ p225) {
  int r = blockIdx.x * blockDim.x + threadIdx.x;
  if (r >= 225) return;
  float ph = (float)(r / 15 - 7);
  float pw = (float)(r % 15 - 7);
  float v[32], u[32];
#pragma unroll
  for (int j = 0; j < 32; ++j) v[j] = ph * w0[j * 2 + 0] + pw * w0[j * 2 + 1] + b0[j];
  ln_relu32(v, g0, bb0);
  for (int j = 0; j < 32; ++j) {
    float acc = b1[j];
    const float* wr = w1 + j * 32;
#pragma unroll
    for (int k = 0; k < 32; ++k) acc += wr[k] * v[k];
    u[j] = acc;
  }
  ln_relu32(u, g1, bb1);
  for (int j = 0; j < 32; ++j) {
    float acc = b2[j];
    const float* wr = w2 + j * 32;
#pragma unroll
    for (int k = 0; k < 32; ++k) acc += wr[k] * u[k];
    v[j] = acc;
  }
  ln_relu32(v, g2, bb2);
  for (int h = 0; h < 16; ++h) {
    float acc = b3[h];
    const float* wr = w3 + h * 32;
#pragma unroll
    for (int k = 0; k < 32; ++k) acc += wr[k] * v[k];
    p225[r * 16 + h] = acc;
  }
}

// ---------------------------------------------------------------------------
// Generic WMMA GEMM:  C[M,N] = A[M,K](f16,row-major) * W[N,K](f16)^T + bias
// Each wave computes a 32x64 tile: 2 A-fragments x 4 B-fragments, 8 WMMA
// per 32-wide K step, all fragment loads 128-bit.
// Epilogue (compile time):
//   0 = QKV  : scatter into [grp][head][pos][d] f16 q/k/v, fold SCALE_Q into q
//   1 = PROJ : un-group + residual add, fp32 into d_out
//   2 = GELU : exact-erf GELU, f16 into mid buffer
//   3 = MLP2 : fp32 += into d_out (which holds x1)
// ---------------------------------------------------------------------------
template <int EPI>
__global__ __launch_bounds__(256)
void gemm_wmma_kernel(const _Float16* __restrict__ A,
                      const _Float16* __restrict__ W,
                      const float* __restrict__ bias,
                      void* __restrict__ outp,
                      const float* __restrict__ resid,
                      int M, int N, int K) {
  int wave = blockIdx.x * (blockDim.x >> 5) + (threadIdx.x >> 5);
  int lane = threadIdx.x & 31;
  int lr = lane & 15;   // row (A) / column (B) selector
  int hi = lane >> 4;   // K-half selector
  int ngroups = N >> 6;
  int mt = wave / ngroups;
  int ng = wave % ngroups;
  int m0 = mt << 5;
  int n0 = ng << 6;

  v8f acc[8];
#pragma unroll
  for (int i = 0; i < 8; ++i) acc[i] = (v8f){};

  const _Float16* Arow0 = A + (size_t)(m0 + lr) * K;
  const _Float16* Arow1 = Arow0 + (size_t)16 * K;
  for (int k0 = 0; k0 < K; k0 += 32) {
    int kb = k0 + (hi << 4);
    v16h a0 = mk16(ld8h(Arow0 + kb), ld8h(Arow0 + kb + 8));
    v16h a1 = mk16(ld8h(Arow1 + kb), ld8h(Arow1 + kb + 8));
    int kw = k0 + (hi << 4);
#pragma unroll
    for (int nb = 0; nb < 4; ++nb) {
      const _Float16* Wr = W + (size_t)(n0 + nb * 16 + lr) * K + kw;
      v16h bf = mk16(ld8h(Wr), ld8h(Wr + 8));
      acc[nb]     = wmma16(a0, bf, acc[nb]);
      acc[nb + 4] = wmma16(a1, bf, acc[nb + 4]);
    }
  }

#pragma unroll
  for (int mi = 0; mi < 2; ++mi) {
#pragma unroll
    for (int nb = 0; nb < 4; ++nb) {
      v8f c = acc[mi * 4 + nb];
#pragma unroll
      for (int r = 0; r < 8; ++r) {
        int row = m0 + mi * 16 + (hi << 3) + r; // C: M = r + 8*(lane>=16)
        int col = n0 + nb * 16 + lr;            //    N = lane & 15
        float val = c[r] + bias[col];
        if constexpr (EPI == 0) {       // QKV scatter into grouped layout
          int t = row;
          int b  = t >> 12;
          int rr = (t >> 6) & 63, cc = t & 63;
          int a = rr >> 3, i = rr & 7, cb = cc >> 3, j = cc & 7;
          int grp = (b << 6) + (i << 3) + j;
          int pos = (a << 3) + cb;
          int s = col >> 9;             // 0=q 1=k 2=v
          int d512 = col & 511;
          int hh = d512 >> 5, dd = d512 & 31;
          if (s == 0) val *= SCALE_Q;
          _Float16* dst = (_Float16*)outp + (size_t)s * 33554432ull +
                          (((size_t)(grp * 16 + hh) * 64 + pos) * 32 + dd);
          *dst = (_Float16)val;
        } else if constexpr (EPI == 1) {  // proj: un-group + residual
          int gr = row;
          int grp = gr >> 6, pos = gr & 63;
          int b = grp >> 6, ij = grp & 63;
          int i = ij >> 3, j = ij & 7;
          int a = pos >> 3, cb = pos & 7;
          int t = (b << 12) + (((a << 3) + i) << 6) + ((cb << 3) + j);
          size_t idx = (size_t)t * DIM_C + col;
          ((float*)outp)[idx] = resid[idx] + val;
        } else if constexpr (EPI == 2) {  // GELU (exact erf) -> f16 mid
          float gv = 0.5f * val * (1.0f + erff(val * 0.70710678118654752f));
          ((_Float16*)outp)[(size_t)row * N + col] = (_Float16)gv;
        } else {                          // MLP2: += into x1 (d_out)
          size_t idx = (size_t)row * DIM_C + col;
          float* o = (float*)outp;
          o[idx] = o[idx] + val;
        }
      }
    }
  }
}

// ---------------------------------------------------------------------------
// Attention: one block of 128 threads (4 waves) per (group, head).
// S = Q K^T (+ dyn pos bias from LDS) via WMMA, fp32 softmax in LDS,
// O = P V via WMMA with V staged transposed in LDS (b128 ds loads).
// ---------------------------------------------------------------------------
__global__ __launch_bounds__(128)
void attn_kernel(const _Float16* __restrict__ q16,
                 const _Float16* __restrict__ k16,
                 const _Float16* __restrict__ v16,
                 const float* __restrict__ p225,
                 _Float16* __restrict__ o16) {
  __shared__ float    S[64][68];   // padded against bank conflicts
  __shared__ _Float16 P[64][72];   // rows 144B-aligned for b128 loads
  __shared__ _Float16 Vt[32][72];  // V transposed: [d][pos]
  __shared__ float    Bh[232];     // per-head pos-bias slice (225 used)

  int g = blockIdx.x >> 4;
  int h = blockIdx.x & 15;
  size_t base = (size_t)(g * 16 + h) * GSIZE * H_DIM;
  const _Float16* Q = q16 + base;
  const _Float16* Km = k16 + base;
  const _Float16* V = v16 + base;

  int wv = threadIdx.x >> 5;
  int lane = threadIdx.x & 31;
  int lr = lane & 15, hi = lane >> 4;

  // ---- stage V transposed + per-head bias slice into LDS ----
  {
    int pos = threadIdx.x >> 1;
    int off = (threadIdx.x & 1) << 4;
    const _Float16* vp = V + (size_t)pos * H_DIM + off;
    v8h u0 = ld8h(vp), u1 = ld8h(vp + 8);
#pragma unroll
    for (int e = 0; e < 8; ++e) {
      Vt[off + e][pos]     = u0[e];
      Vt[off + 8 + e][pos] = u1[e];
    }
    for (int i = threadIdx.x; i < 225; i += 128) Bh[i] = p225[i * 16 + h];
  }
  __syncthreads();

  // ---- phase 1: S = Q K^T + bias (K dim = 32, single WMMA step) ----
  {
    const _Float16* qrow = Q + (size_t)(16 * wv + lr) * H_DIM;
    v16h af = mk16(ld8h(qrow + (hi << 4)), ld8h(qrow + (hi << 4) + 8));
#pragma unroll
    for (int nb = 0; nb < 4; ++nb) {
      const _Float16* krow = Km + (size_t)(nb * 16 + lr) * H_DIM + (hi << 4);
      v16h bf = mk16(ld8h(krow), ld8h(krow + 8));
      v8f c = {};
      c = wmma16(af, bf, c);
#pragma unroll
      for (int r = 0; r < 8; ++r) {
        int qn = 16 * wv + (hi << 3) + r;   // query position
        int km = nb * 16 + lr;              // key position
        int a1 = qn >> 3, c1 = qn & 7, a2 = km >> 3, c2 = km & 7;
        int ridx = (a1 - a2 + 7) * 15 + (c1 - c2 + 7);
        S[qn][km] = c[r] + Bh[ridx];
      }
    }
  }
  __syncthreads();

  // ---- phase 2: softmax, 2 threads per row (32 cols each), pair via shfl --
  {
    int t = threadIdx.x >> 1;      // row
    int c0 = (threadIdx.x & 1) << 5;
    float mx = -3.4e38f;
    for (int m = 0; m < 32; ++m) mx = fmaxf(mx, S[t][c0 + m]);
    mx = fmaxf(mx, __shfl_xor(mx, 1, 32));
    float sum = 0.f;
    for (int m = 0; m < 32; ++m) {
      float e = __expf(S[t][c0 + m] - mx);
      S[t][c0 + m] = e;
      sum += e;
    }
    sum += __shfl_xor(sum, 1, 32);
    float inv = 1.f / sum;
    for (int m = 0; m < 32; ++m) P[t][c0 + m] = (_Float16)(S[t][c0 + m] * inv);
  }
  __syncthreads();

  // ---- phase 3: O = P V  (K dim = 64 -> 2 WMMA steps, N = 32 -> 2 tiles) --
  v8f oc0 = {}, oc1 = {};
#pragma unroll
  for (int k0 = 0; k0 < 64; k0 += 32) {
    const _Float16* prow = &P[16 * wv + lr][0] + k0 + (hi << 4);
    v16h pa = mk16(ld8h(prow), ld8h(prow + 8));
#pragma unroll
    for (int nb = 0; nb < 2; ++nb) {
      const _Float16* vr = &Vt[nb * 16 + lr][0] + k0 + (hi << 4);
      v16h bf = mk16(ld8h(vr), ld8h(vr + 8));
      v8f c = (nb == 0) ? oc0 : oc1;
      c = wmma16(pa, bf, c);
      if (nb == 0) oc0 = c; else oc1 = c;
    }
  }
#pragma unroll
  for (int nb = 0; nb < 2; ++nb) {
    v8f c = (nb == 0) ? oc0 : oc1;
#pragma unroll
    for (int r = 0; r < 8; ++r) {
      int m = 16 * wv + (hi << 3) + r;
      int dcol = nb * 16 + lr;
      o16[(size_t)(g * 64 + m) * DIM_C + h * 32 + dcol] = (_Float16)c[r];
    }
  }
}

// ---------------------------------------------------------------------------
// host launcher
// ---------------------------------------------------------------------------
extern "C" void kernel_launch(void* const* d_in, const int* in_sizes, int n_in,
                              void* d_out, int out_size, void* d_ws, size_t ws_size,
                              hipStream_t stream) {
  const float* x        = (const float*)d_in[0];
  const float* norm1_g  = (const float*)d_in[1];
  const float* norm1_b  = (const float*)d_in[2];
  const float* qkv_w    = (const float*)d_in[3];
  const float* qkv_b    = (const float*)d_in[4];
  const float* proj_w   = (const float*)d_in[5];
  const float* proj_b   = (const float*)d_in[6];
  const float* pos_w0   = (const float*)d_in[7];
  const float* pos_b0   = (const float*)d_in[8];
  const float* pos_g0   = (const float*)d_in[9];
  const float* pos_bb0  = (const float*)d_in[10];
  const float* pos_w1   = (const float*)d_in[11];
  const float* pos_b1   = (const float*)d_in[12];
  const float* pos_g1   = (const float*)d_in[13];
  const float* pos_bb1  = (const float*)d_in[14];
  const float* pos_w2   = (const float*)d_in[15];
  const float* pos_b2   = (const float*)d_in[16];
  const float* pos_g2   = (const float*)d_in[17];
  const float* pos_bb2  = (const float*)d_in[18];
  const float* pos_w3   = (const float*)d_in[19];
  const float* pos_b3   = (const float*)d_in[20];
  const float* norm2_g  = (const float*)d_in[21];
  const float* norm2_b  = (const float*)d_in[22];
  const float* mlp_w1   = (const float*)d_in[23];
  const float* mlp_b1   = (const float*)d_in[24];
  const float* mlp_w2   = (const float*)d_in[25];
  const float* mlp_b2   = (const float*)d_in[26];

  char* ws = (char*)d_ws;
  _Float16* h16    = (_Float16*)(ws + H16_OFF);
  _Float16* q16    = (_Float16*)(ws + Q16_OFF);
  _Float16* k16    = (_Float16*)(ws + K16_OFF);
  _Float16* v16    = (_Float16*)(ws + V16_OFF);
  _Float16* o16    = (_Float16*)(ws + O16_OFF);
  _Float16* mid16  = (_Float16*)(ws + MID16_OFF);
  float*    p225   = (float*)   (ws + P225_OFF);
  _Float16* qkvw16 = (_Float16*)(ws + QKVW_OFF);
  _Float16* projw16= (_Float16*)(ws + PROJW_OFF);
  _Float16* w1_16  = (_Float16*)(ws + W1_OFF);
  _Float16* w2_16  = (_Float16*)(ws + W2_OFF);

  // weight conversions
  cvt_f16_kernel<<<(1536 * 512 + 255) / 256, 256, 0, stream>>>(qkv_w, qkvw16, 1536 * 512);
  cvt_f16_kernel<<<(512 * 512 + 255) / 256, 256, 0, stream>>>(proj_w, projw16, 512 * 512);
  cvt_f16_kernel<<<(2048 * 512 + 255) / 256, 256, 0, stream>>>(mlp_w1, w1_16, 2048 * 512);
  cvt_f16_kernel<<<(512 * 2048 + 255) / 256, 256, 0, stream>>>(mlp_w2, w2_16, 512 * 2048);

  // LN1 -> h16
  ln_f16_kernel<<<N_TOK / 8, 256, 0, stream>>>(x, norm1_g, norm1_b, h16, N_TOK);

  // dynamic position bias table
  pos_mlp_kernel<<<1, 256, 0, stream>>>(pos_w0, pos_b0, pos_g0, pos_bb0,
                                        pos_w1, pos_b1, pos_g1, pos_bb1,
                                        pos_w2, pos_b2, pos_g2, pos_bb2,
                                        pos_w3, pos_b3, p225);

  // QKV GEMM: M=65536, N=1536, K=512; 32x64 tile/wave
  gemm_wmma_kernel<0><<<(N_TOK / 32) * (1536 / 64) / 8, 256, 0, stream>>>(
      h16, qkvw16, qkv_b, (void*)q16, nullptr, N_TOK, 1536, 512);

  // attention: 1024 groups x 16 heads
  attn_kernel<<<N_GRP * N_HEAD, 128, 0, stream>>>(q16, k16, v16, p225, o16);

  // proj GEMM + un-group + residual -> d_out (x1)
  gemm_wmma_kernel<1><<<(N_TOK / 32) * (512 / 64) / 8, 256, 0, stream>>>(
      o16, projw16, proj_b, d_out, x, N_TOK, 512, 512);

  // LN2(x1) -> h16 (reused)
  ln_f16_kernel<<<N_TOK / 8, 256, 0, stream>>>((const float*)d_out, norm2_g,
                                               norm2_b, h16, N_TOK);

  // MLP1 + GELU -> mid16
  gemm_wmma_kernel<2><<<(N_TOK / 32) * (2048 / 64) / 8, 256, 0, stream>>>(
      h16, w1_16, mlp_b1, (void*)mid16, nullptr, N_TOK, 2048, 512);

  // MLP2 += into d_out
  gemm_wmma_kernel<3><<<(N_TOK / 32) * (512 / 64) / 8, 256, 0, stream>>>(
      mid16, w2_16, mlp_b2, d_out, nullptr, N_TOK, 512, 2048);
}
